// MLAttention_1683627180889
// MI455X (gfx1250) — matrix operand order repaired and checked by reading, not compiled, not run on hardware
//
#include <hip/hip_runtime.h>
#include <stdint.h>

constexpr int BB = 16;    // batch
constexpr int SS = 512;   // seq
constexpr int HH = 768;   // hidden
constexpr int LL = 4096;  // out features

typedef __attribute__((ext_vector_type(16))) __bf16   v16bf;
typedef __attribute__((ext_vector_type(8)))  float    v8f;
typedef __attribute__((ext_vector_type(4)))  uint32_t v4u;
typedef __attribute__((ext_vector_type(2)))  uint32_t v2u;
typedef __attribute__((ext_vector_type(4)))  float    v4f;
typedef __attribute__((ext_vector_type(4)))  uint8_t  v4b;
typedef __attribute__((ext_vector_type(4)))  uint32_t u32x4;
typedef __attribute__((ext_vector_type(8)))  int      i32x8;
typedef __attribute__((ext_vector_type(4)))  int      i32x4;

union BF16x16 { v16bf v; v4u q[2]; uint32_t u[8]; };

#if __has_builtin(__builtin_amdgcn_tensor_load_to_lds)
#define HAVE_TDM 1
#else
#define HAVE_TDM 0
#endif

__device__ __forceinline__ uint16_t f2bf(float f) {
  uint32_t b = __builtin_bit_cast(uint32_t, f);
  return (uint16_t)((b + 0x7FFFu + ((b >> 16) & 1u)) >> 16);  // RNE
}
__device__ __forceinline__ uint32_t pk2(float lo, float hi) {
  return (uint32_t)f2bf(lo) | ((uint32_t)f2bf(hi) << 16);
}

// ---- stage one 64-row x 32-elem bf16 tile (4 KB) into LDS -----------------
// TDM path: 2-D Tensor DMA descriptor per CDNA5 ISA
// tracked by TENSORcnt. Fallback: cooperative per-lane copy.
// dstLdsAddr is the raw LDS byte offset (kept as integer math so the DS read
// pointers never mix with ptrtoint and stay in address space 3).
__device__ __forceinline__ void stage_tile(uint16_t* dstPtr, uint32_t dstLdsAddr,
                                           const uint16_t* src,
                                           uint32_t tensD0, uint32_t tensD1,
                                           uint32_t strideD0, int lane) {
#if HAVE_TDM
  uint64_t ga = (uint64_t)(uintptr_t)src;
  u32x4 g0;
  g0[0] = 1u;                                   // count=1, load, user D#
  g0[1] = dstLdsAddr;                           // lds_addr (bytes)
  g0[2] = (uint32_t)ga;                         // global_addr[31:0]
  g0[3] = (uint32_t)((ga >> 32) & 0x01FFFFFFu)  // global_addr[56:32]
          | (2u << 30);                         // type = 2 ("image")
  i32x8 g1;
  g1[0] = (int)(1u << 16);                      // wg_mask=0, data_size=1 (2 bytes)
  g1[1] = (int)((tensD0 & 0xFFFFu) << 16);      // tensor_dim0[15:0]
  g1[2] = (int)((tensD0 >> 16) | ((tensD1 & 0xFFFFu) << 16));
  g1[3] = (int)((tensD1 >> 16) | (32u << 16));  // tile_dim0 = 32 elems
  g1[4] = (int)64u;                             // tile_dim1 = 64 rows, tile_dim2 = 0
  g1[5] = (int)strideD0;                        // tensor_dim0_stride[31:0]
  g1[6] = 0;                                    // stride0[47:32]=0, stride1 lo=0
  g1[7] = 0;
  i32x4 z4 = {0, 0, 0, 0};
#if __has_include(<hip/amd_detail/amd_gfx1250_TDM.h>)
  i32x8 z8 = {0, 0, 0, 0, 0, 0, 0, 0};
  __builtin_amdgcn_tensor_load_to_lds(g0, g1, z4, z4, z8, 0);
#else
  __builtin_amdgcn_tensor_load_to_lds(g0, g1, z4, z4, 0);
#endif
  (void)dstPtr; (void)lane;
#else
  for (int r = lane; r < 64; r += 32) {         // 2 rows / lane, 64 B each
    const v4u* s = (const v4u*)(src + (size_t)r * strideD0);
    v4u* d = (v4u*)(dstPtr + r * 32);
    d[0] = s[0];
    d[1] = s[1];
  }
  (void)tensD0; (void)tensD1; (void)dstLdsAddr;
#endif
}

__device__ __forceinline__ void stage_wait_pending1() {
#if HAVE_TDM
  __builtin_amdgcn_s_wait_tensorcnt((short)1);
#endif
  asm volatile("" ::: "memory");
}
__device__ __forceinline__ void stage_wait_all() {
#if HAVE_TDM
  __builtin_amdgcn_s_wait_tensorcnt((short)0);
#endif
  asm volatile("" ::: "memory");
}

// ---- prep: W f32 -> bf16 -------------------------------------------------
__global__ void cvt_w_kernel(const float* __restrict__ W, uint16_t* __restrict__ Wb) {
  int i = blockIdx.x * 256 + threadIdx.x;       // exact grid
  Wb[i] = f2bf(W[i]);
}

// ---- prep: X f32 -> bf16 (row-major for GEMM1) + transposed copy (GEMM2) --
__global__ void cvt_x_kernel(const float* __restrict__ X,
                             uint16_t* __restrict__ Xb,
                             uint16_t* __restrict__ XT) {
  int i = blockIdx.x * 256 + threadIdx.x;       // exact grid
  int h  = i % HH;
  int bs = i / HH;
  int s  = bs % SS;
  int b  = bs / SS;
  uint16_t v = f2bf(X[i]);
  Xb[i] = v;
  XT[((size_t)b * HH + h) * SS + s] = v;
}

// LDS layout (dynamic): mask 512 B | scores f32 4x32 KB | P bf16 4x16 KB | stage 4x8 KB
constexpr int LDS_SCORE = 512;
constexpr int LDS_PBF   = LDS_SCORE + 4 * 16 * SS * 4;   // 131584
constexpr int LDS_STAGE = LDS_PBF + 4 * 16 * SS * 2;     // 197120
constexpr int LDS_TOTAL = LDS_STAGE + 4 * 2 * 4096;      // 229888

// ---- fused scores -> masked softmax -> weighted pooling -------------------
__global__ __launch_bounds__(128) void attn_pool_kernel(
    const uint16_t* __restrict__ Wb,   // [L,H] bf16
    const uint16_t* __restrict__ Xb,   // [B,S,H] bf16
    const uint16_t* __restrict__ XT,   // [B,H,S] bf16
    const uint8_t*  __restrict__ mask, // [B,S] bool
    float* __restrict__ out)           // [B,L,H] f32
{
  extern __shared__ char smem[];
  uint8_t* mLds = (uint8_t*)smem;
  const int wave = threadIdx.x >> 5;
  const int lane = threadIdx.x & 31;
  const int m    = lane & 15;   // A row / B,C,D column
  const int hv   = lane >> 4;   // half-wave select
  const int b    = blockIdx.y;
  const int l0   = blockIdx.x * 64 + wave * 16;

  float*    scoreLds = (float*)(smem + LDS_SCORE) + wave * (16 * SS);
  uint16_t* Pbf      = (uint16_t*)(smem + LDS_PBF) + wave * (16 * SS);
  uint16_t* stgBase  = (uint16_t*)(smem + LDS_STAGE + wave * 8192);   // 2x4 KB ring
  const uint32_t stgLds0 =
      (uint32_t)(uintptr_t)smem + (uint32_t)(LDS_STAGE + wave * 8192);

  const v8f vzero = {0.f, 0.f, 0.f, 0.f, 0.f, 0.f, 0.f, 0.f};

  // stage this batch's mask (512 bytes) into LDS
  ((uint32_t*)mLds)[threadIdx.x] =
      ((const uint32_t*)(mask + (size_t)b * SS))[threadIdx.x];
  __syncthreads();

  // ===== Phase 1: scores[16,512] = W_tile(16xH) x X[b]^T(HxS) ==============
  // Flattened tile loop t = sb*24 + hc; TDM double-buffers the B tile
  // (X[b, s0:s0+64, h0:h0+32]), registers double-buffer the A tile (W rows).
  {
    const uint16_t* xbase = Xb + (size_t)b * SS * HH;
    stage_tile(stgBase, stgLds0, xbase, HH, SS, HH, lane);     // (sb=0,hc=0)
    BF16x16 Acur;
    {
      const uint16_t* wrow = Wb + (size_t)(l0 + m) * HH + hv * 8;  // hc=0
      Acur.q[0] = *(const v4u*)(wrow);
      Acur.q[1] = *(const v4u*)(wrow + 16);
    }
    v8f acc[4] = {vzero, vzero, vzero, vzero};
    for (int t = 0; t < 192; ++t) {
      const int hc = t % 24;
      BF16x16 Anext;
      if (t < 191) {
        const int nt = t + 1, nsb = nt / 24, nhc = nt % 24;
        stage_tile(stgBase + (nt & 1) * 2048, stgLds0 + (nt & 1) * 4096,
                   xbase + (size_t)(nsb * 64) * HH + nhc * 32, HH, SS, HH, lane);
        const uint16_t* wrow = Wb + (size_t)(l0 + m) * HH + nhc * 32 + hv * 8;
        Anext.q[0] = *(const v4u*)(wrow);
        Anext.q[1] = *(const v4u*)(wrow + 16);
        stage_wait_pending1();                            // tile t complete
      } else {
        stage_wait_all();
      }
      const uint16_t* sbuf = stgBase + (t & 1) * 2048;    // pure AS3 arithmetic
      BF16x16 Bt[4];
#pragma unroll
      for (int j = 0; j < 4; ++j) {                       // one 8-load DS clause
        const uint16_t* bp = sbuf + (j * 16 + m) * 32 + hv * 8;
        Bt[j].q[0] = *(const v4u*)(bp);
        Bt[j].q[1] = *(const v4u*)(bp + 16);
      }
#pragma unroll
      for (int j = 0; j < 4; ++j)
        acc[j] = __builtin_amdgcn_wmma_f32_16x16x32_bf16(
            false, Acur.v, false, Bt[j].v, (short)0, acc[j], false, false);
      if (t < 191) Acur = Anext;
      if (hc == 23) {                                     // K loop done: spill strip
        const int s0 = (t / 24) * 64;
#pragma unroll
        for (int j = 0; j < 4; ++j)
#pragma unroll
          for (int v = 0; v < 8; ++v)                     // D: row v+8*hv, col m
            scoreLds[(v + 8 * hv) * SS + s0 + j * 16 + m] = acc[j][v];
        acc[0] = vzero; acc[1] = vzero; acc[2] = vzero; acc[3] = vzero;
      }
    }
  }
  __syncthreads();

  // ===== Phase 2: masked softmax over S; emit P as bf16 ====================
  // Sweep 1: masked row max.  Sweep 2: e=exp(s-mx) stored in place + sum.
  // Sweep 3: normalize + pack bf16.  (exp evaluated once per element)
  {
    const int    cb   = hv * 256;
    float*       rowp = scoreLds + m * SS + cb;
    const float  NEG  = -3.402823466e38f;
    float mx = NEG;
    for (int c = 0; c < 256; c += 4) {
      v4f sv = *(const v4f*)(rowp + c);
      v4b mk = *(const v4b*)(mLds + cb + c);
      mx = fmaxf(mx, mk.x ? sv.x : NEG);
      mx = fmaxf(mx, mk.y ? sv.y : NEG);
      mx = fmaxf(mx, mk.z ? sv.z : NEG);
      mx = fmaxf(mx, mk.w ? sv.w : NEG);
    }
    mx = fmaxf(mx, __shfl_xor(mx, 16, 32));
    float sum = 0.f;
    for (int c = 0; c < 256; c += 4) {
      v4f sv = *(const v4f*)(rowp + c);
      v4b mk = *(const v4b*)(mLds + cb + c);
      v4f ev;
      ev.x = mk.x ? __expf(sv.x - mx) : 0.f;
      ev.y = mk.y ? __expf(sv.y - mx) : 0.f;
      ev.z = mk.z ? __expf(sv.z - mx) : 0.f;
      ev.w = mk.w ? __expf(sv.w - mx) : 0.f;
      sum += ev.x + ev.y + ev.z + ev.w;
      *(v4f*)(rowp + c) = ev;                             // in-place, same index
    }
    sum += __shfl_xor(sum, 16, 32);
    const float inv = 1.f / sum;
    uint16_t* prow = Pbf + m * SS + cb;
    for (int c = 0; c < 256; c += 4) {
      v4f ev = *(const v4f*)(rowp + c);
      v2u st = {pk2(ev.x * inv, ev.y * inv), pk2(ev.z * inv, ev.w * inv)};
      *(v2u*)(prow + c) = st;
    }
  }
  __syncthreads();

  // ===== Phase 3: out[16,H] = P(16xS) x X[b](SxH) ==========================
  // Flattened tile loop t = hb*16 + sc; TDM double-buffers the B tile
  // (XT[b, hbase:hbase+64, s0:s0+32]); A = bf16 probabilities from LDS.
  {
    const uint16_t* xtb = XT + (size_t)b * HH * SS;
    stage_tile(stgBase, stgLds0, xtb, SS, HH, SS, lane);       // (hb=0,sc=0)
    v8f acc[4] = {vzero, vzero, vzero, vzero};
    for (int t = 0; t < 192; ++t) {
      const int hb = t >> 4, sc = t & 15;
      if (t < 191) {
        const int nt = t + 1, nhb = nt >> 4, nsc = nt & 15;
        stage_tile(stgBase + (nt & 1) * 2048, stgLds0 + (nt & 1) * 4096,
                   xtb + (size_t)(nhb * 64) * SS + nsc * 32, SS, HH, SS, lane);
        stage_wait_pending1();
      } else {
        stage_wait_all();
      }
      BF16x16 A;                                          // P rows from LDS
      const uint16_t* ap = Pbf + m * SS + sc * 32 + hv * 8;
      A.q[0] = *(const v4u*)(ap);
      A.q[1] = *(const v4u*)(ap + 16);
      const uint16_t* sbuf = stgBase + (t & 1) * 2048;
      BF16x16 Bt[4];
#pragma unroll
      for (int j = 0; j < 4; ++j) {
        const uint16_t* bp = sbuf + (j * 16 + m) * 32 + hv * 8;
        Bt[j].q[0] = *(const v4u*)(bp);
        Bt[j].q[1] = *(const v4u*)(bp + 16);
      }
#pragma unroll
      for (int j = 0; j < 4; ++j)
        acc[j] = __builtin_amdgcn_wmma_f32_16x16x32_bf16(
            false, A.v, false, Bt[j].v, (short)0, acc[j], false, false);
      if (sc == 15) {                                     // K loop done: write out
        const int hbase = hb * 64;
#pragma unroll
        for (int j = 0; j < 4; ++j)
#pragma unroll
          for (int v = 0; v < 8; ++v)
            out[((size_t)b * LL + l0 + v + 8 * hv) * HH + hbase + j * 16 + m] =
                acc[j][v];
        acc[0] = vzero; acc[1] = vzero; acc[2] = vzero; acc[3] = vzero;
      }
    }
  }
}

extern "C" void kernel_launch(void* const* d_in, const int* in_sizes, int n_in,
                              void* d_out, int out_size, void* d_ws, size_t ws_size,
                              hipStream_t stream) {
  const float*   X    = (const float*)d_in[0];    // [B,S,H] f32
  const uint8_t* mask = (const uint8_t*)d_in[1];  // [B,S] bool
  const float*   W    = (const float*)d_in[2];    // [L,H] f32
  float*         out  = (float*)d_out;            // [B,L,H] f32

  uint8_t*  ws = (uint8_t*)d_ws;
  uint16_t* Wb = (uint16_t*)(ws);
  uint16_t* Xb = (uint16_t*)(ws + (size_t)LL * HH * 2);
  uint16_t* XT = (uint16_t*)(ws + (size_t)LL * HH * 2 + (size_t)BB * SS * HH * 2);

  cvt_w_kernel<<<(LL * HH) / 256, 256, 0, stream>>>(W, Wb);
  cvt_x_kernel<<<(BB * SS * HH) / 256, 256, 0, stream>>>(X, Xb, XT);

  attn_pool_kernel<<<dim3(LL / 64, BB), 128, LDS_TOTAL, stream>>>(
      Wb, Xb, XT, mask, out);
}
